// Decoder_76046690943024
// MI455X (gfx1250) — matrix operand order repaired and checked by reading, not compiled, not run on hardware
//
#include <hip/hip_runtime.h>
#include <hip/hip_bf16.h>

// ---------------------------------------------------------------------------
// Social-trajectory decoder for MI455X (gfx1250, wave32, WMMA).
//
// Dominant cost: pairwise pooling, rewritten as
//   h1[i,j,:] = relu(A[i,:] + B[j,:])      (rank-structured first layer)
//   h2[i,j,:] = relu(h1 @ W2 + b2)         (N^2 x [64x64] GEMM -> WMMA f16)
//   ctx[i,:]  = max over valid j (init 0)  (h2 >= 0 makes -inf trick = max w/ 0)
//
// k_pool v2: each wave owns a j-chunk and computes ALL 4 output column tiles
// (8 WMMAs per A-fragment build) instead of 4 waves redundantly building the
// same A-fragment for 2 WMMAs each.
// ---------------------------------------------------------------------------

#define NPED 512
#define TSTEPS 12
#define HID 64
#define CTXD 64
#define LSTM_IN 128
#define EMB_IN 162   // 2 + 64 + 32 + 64

typedef __attribute__((ext_vector_type(16))) _Float16 v16h;
typedef __attribute__((ext_vector_type(8)))  float    v8f;

__device__ __forceinline__ float sigm(float x) { return 1.0f / (1.0f + expf(-x)); }

// ---------------------------------------------------------------------------
// init: copy initial state into workspace
// ---------------------------------------------------------------------------
__global__ void k_init(const float* __restrict__ lastpos,
                       const float* __restrict__ obs,
                       const float* __restrict__ h0,
                       const float* __restrict__ c0,
                       float* H, float* CS, float* CTX, float* CUR, float* PRV) {
    int tid = blockIdx.x * blockDim.x + threadIdx.x;
    if (tid < NPED * HID) {
        H[tid]   = h0[tid];
        CS[tid]  = c0[tid];
        CTX[tid] = 0.0f;
    }
    if (tid < NPED * 2) {
        CUR[tid] = obs[7 * NPED * 2 + tid];  // obs_traj_pos[-1]
        PRV[tid] = lastpos[tid];
    }
}

// ---------------------------------------------------------------------------
// precompute: M2 = W_sp @ W1[0:32]  (2x64), K0 = b_sp @ W1[0:32] + b1  (64)
// ---------------------------------------------------------------------------
__global__ void k_pre(const float* __restrict__ W_sp, const float* __restrict__ b_sp,
                      const float* __restrict__ W1,   const float* __restrict__ b1,
                      float* M2, float* K0) {
    int k = threadIdx.x;  // 0..63
    float m0 = 0.f, m1 = 0.f, k0 = b1[k];
    for (int e = 0; e < 32; ++e) {
        float w = W1[e * 64 + k];
        m0 += W_sp[e]      * w;   // W_sp row 0
        m1 += W_sp[32 + e] * w;   // W_sp row 1
        k0 += b_sp[e]      * w;
    }
    M2[k] = m0; M2[64 + k] = m1; K0[k] = k0;
}

// ---------------------------------------------------------------------------
// LSTM step: emb -> relu MLP -> gates -> (h, c). One block (256 thr) per agent.
// ---------------------------------------------------------------------------
__global__ void __launch_bounds__(256)
k_lstm(const float* __restrict__ PRV, const float* __restrict__ cond,
       const float* __restrict__ z,   const float* __restrict__ CTX,
       float* H, float* CS,
       const float* __restrict__ W_in, const float* __restrict__ b_in,
       const float* __restrict__ W_ih, const float* __restrict__ W_hh,
       const float* __restrict__ b_ih, const float* __restrict__ b_hh) {
    __shared__ float emb[EMB_IN];
    __shared__ float xv[LSTM_IN];
    __shared__ float gates[4 * HID];
    __shared__ float hloc[HID];
    const int p = blockIdx.x, tid = threadIdx.x;

    if (tid < 2)        emb[tid] = PRV[p * 2 + tid];
    else if (tid < 66)  emb[tid] = cond[p * 64 + (tid - 2)];
    else if (tid < 98)  emb[tid] = z[p * 32 + (tid - 66)];
    else if (tid < 162) emb[tid] = CTX[p * 64 + (tid - 98)];
    if (tid >= 192)     hloc[tid - 192] = H[p * 64 + (tid - 192)];
    __syncthreads();

    if (tid < LSTM_IN) {
        float a = b_in[tid];
        for (int k = 0; k < EMB_IN; ++k) a += emb[k] * W_in[k * LSTM_IN + tid];
        xv[tid] = fmaxf(a, 0.0f);
    }
    __syncthreads();

    {
        float g = b_ih[tid] + b_hh[tid];
        for (int k = 0; k < LSTM_IN; ++k) g += xv[k] * W_ih[tid * LSTM_IN + k];
        for (int q = 0; q < HID; ++q)     g += hloc[q] * W_hh[tid * HID + q];
        gates[tid] = g;
    }
    __syncthreads();

    if (tid < HID) {
        float ig = gates[tid], fg = gates[64 + tid];
        float gg = gates[128 + tid], og = gates[192 + tid];
        float cs = CS[p * 64 + tid];
        float cn = sigm(fg) * cs + sigm(ig) * tanhf(gg);
        float hn = sigm(og) * tanhf(cn);
        CS[p * 64 + tid] = cn;
        H[p * 64 + tid]  = hn;
    }
}

// ---------------------------------------------------------------------------
// A/B vectors of the rank-structured first pooling layer. 64 threads / agent.
//   A[p,k] =  pos_p.M2[:,k] + h_p . W1[96:160, k]
//   B[p,k] = -pos_p.M2[:,k] + h_p . W1[32:96,  k] + K0[k]
// ---------------------------------------------------------------------------
__global__ void __launch_bounds__(64)
k_ab(const float* __restrict__ H, const float* __restrict__ CUR,
     const float* __restrict__ W1, const float* __restrict__ M2,
     const float* __restrict__ K0, float* AV, float* BV) {
    __shared__ float hloc[HID];
    __shared__ float pos[2];
    const int p = blockIdx.x, k = threadIdx.x;
    hloc[k] = H[p * 64 + k];
    if (k < 2) pos[k] = CUR[p * 2 + k];
    __syncthreads();
    float pm = pos[0] * M2[k] + pos[1] * M2[64 + k];
    float ai = 0.f, bj = 0.f;
    for (int q = 0; q < HID; ++q) {
        float hq = hloc[q];
        ai += hq * W1[(96 + q) * 64 + k];
        bj += hq * W1[(32 + q) * 64 + k];
    }
    AV[p * 64 + k] = pm + ai;
    BV[p * 64 + k] = -pm + bj + K0[k];
}

// ---------------------------------------------------------------------------
// Pooling with WMMA. One block per row i; 4 waves. Wave w owns j-chunks
// {ob*64 + 16w : ob in 0..7} and computes all 4 output column tiles:
//   - builds its A-fragment (16x32 f16, rows = j_local) ONCE per chunk,
//   - 8 accumulating v_wmma_f32_16x16x32_f16 against resident W2 fragments,
//   - C initialized with broadcast b2 (D column = lane -> uniform bias),
//   - relu + neighbor mask (one v_cmp per row, shared by all 4 tiles) + max.
// Final 4wave x 2half max-reduce through LDS.
// ---------------------------------------------------------------------------
__global__ void __launch_bounds__(128)
k_pool(const float* __restrict__ AV, const float* __restrict__ BV,
       const int* __restrict__ nei_t, const float* __restrict__ W2,
       const float* __restrict__ b2, float* CTX) {
    __shared__ int   neirow[NPED];       // 2 KB
    __shared__ float bch[64 * 64];       // 16 KB: 64 j-rows per outer iter
    __shared__ float avloc[64];
    __shared__ float pred[4][4][32];     // [wave][ntile][lane]

    const int i = blockIdx.x, tid = threadIdx.x;
    const int wave = tid >> 5, lane = tid & 31;
    const int m = lane & 15, hi = lane >> 4;

    // stage neighbor row (4 ints / thread, coalesced) + A vector
    {
        const int base = tid * 4;
        const int* src = nei_t + (size_t)i * NPED + base;
#pragma unroll
        for (int e = 0; e < 4; ++e) neirow[base + e] = src[e];
    }
    if (tid < 64) avloc[tid] = AV[i * 64 + tid];
    __syncthreads();

    // Resident W2 fragments for ALL 4 column tiles (f32 -> f16).
    // 16-bit B layout: lane = column, K = (v/4)*16 + hi*8 + (v%4)*2 (+1)
    // (same as A); second K-step adds 32.
    v16h bf0[4], bf1[4];
    float b2v[4];
#pragma unroll
    for (int nt = 0; nt < 4; ++nt) {
        const int ncol = nt * 16 + m;
        b2v[nt] = b2[ncol];
#pragma unroll
        for (int v = 0; v < 8; ++v) {
            int kb = ((v >> 2) << 4) + hi * 8 + ((v & 3) << 1);
            bf0[nt][2 * v]     = (_Float16)W2[(kb)      * 64 + ncol];
            bf0[nt][2 * v + 1] = (_Float16)W2[(kb + 1)  * 64 + ncol];
            bf1[nt][2 * v]     = (_Float16)W2[(kb + 32) * 64 + ncol];
            bf1[nt][2 * v + 1] = (_Float16)W2[(kb + 33) * 64 + ncol];
        }
    }

    // Per-lane A[i,k] at this lane's A-fragment K positions
    // (16-bit A layout: K = (v/4)*16 + hi*8 + (v%4)*2 (+1), +32 for k-step 1).
    float av0[16], av1[16];
#pragma unroll
    for (int v = 0; v < 8; ++v) {
        int kb = ((v >> 2) << 4) + hi * 8 + ((v & 3) << 1);
        av0[2 * v]     = avloc[kb];
        av0[2 * v + 1] = avloc[kb + 1];
        av1[2 * v]     = avloc[kb + 32];
        av1[2 * v + 1] = avloc[kb + 33];
    }

    float mx[4] = {0.f, 0.f, 0.f, 0.f};  // h2 >= 0 => max-with-0 == -inf trick

    for (int ob = 0; ob < 8; ++ob) {
        const int jbase = ob * 64;
        __syncthreads();
        // cooperative stage: 64 rows x 64 floats; 32 contiguous floats/thread
        {
            const float* srcb = BV + jbase * 64 + tid * 32;
            float* dstb = &bch[tid * 32];
#pragma unroll
            for (int e = 0; e < 32; ++e) dstb[e] = srcb[e];
        }
        __syncthreads();

        const int j0 = jbase + wave * 16;           // this wave's chunk
        const float* brow = &bch[(wave * 16 + m) * 64];

        v16h a0, a1;
#pragma unroll
        for (int v = 0; v < 8; ++v) {
            int kb = ((v >> 2) << 4) + hi * 8 + ((v & 3) << 1);
            a0[2 * v]     = (_Float16)fmaxf(av0[2 * v]     + brow[kb],      0.f);
            a0[2 * v + 1] = (_Float16)fmaxf(av0[2 * v + 1] + brow[kb + 1],  0.f);
            a1[2 * v]     = (_Float16)fmaxf(av1[2 * v]     + brow[kb + 32], 0.f);
            a1[2 * v + 1] = (_Float16)fmaxf(av1[2 * v + 1] + brow[kb + 33], 0.f);
        }

        v8f acc[4];
#pragma unroll
        for (int nt = 0; nt < 4; ++nt) {
#pragma unroll
            for (int r = 0; r < 8; ++r) acc[nt][r] = b2v[nt];  // bias as C
            acc[nt] = __builtin_amdgcn_wmma_f32_16x16x32_f16(
                false, a0, false, bf0[nt], (short)0, acc[nt], false, false);
            acc[nt] = __builtin_amdgcn_wmma_f32_16x16x32_f16(
                false, a1, false, bf1[nt], (short)0, acc[nt], false, false);
        }

        // D layout: lane hi-half holds rows hi*8 + r; mask shared across tiles
#pragma unroll
        for (int r = 0; r < 8; ++r) {
            const bool nv = neirow[j0 + hi * 8 + r] > 0;
#pragma unroll
            for (int nt = 0; nt < 4; ++nt) {
                float h2 = fmaxf(acc[nt][r], 0.0f);
                mx[nt] = nv ? fmaxf(mx[nt], h2) : mx[nt];
            }
        }
    }

#pragma unroll
    for (int nt = 0; nt < 4; ++nt) pred[wave][nt][lane] = mx[nt];
    __syncthreads();
    if (tid < 64) {
        const int nt = tid >> 4, n = tid & 15;
        float f = 0.f;
#pragma unroll
        for (int w = 0; w < 4; ++w)
            f = fmaxf(f, fmaxf(pred[w][nt][n], pred[w][nt][n + 16]));
        CTX[i * 64 + nt * 16 + n] = f;
    }
}

// ---------------------------------------------------------------------------
// mu / logvar / pos projection; update cur_abs & prev_pos; emit outputs.
// ---------------------------------------------------------------------------
__global__ void __launch_bounds__(256)
k_out(const float* __restrict__ H, const float* __restrict__ CTX,
      const float* __restrict__ eps_t, const float* __restrict__ W_l2p,
      const float* __restrict__ b_l2p,
      float* PRV, float* CUR,
      float* out_pos, float* out_mu, float* out_lv) {
    const int p = blockIdx.x * blockDim.x + threadIdx.x;
    if (p >= NPED) return;
    const float* hr = H + p * 64;
    const float* cr = CTX + p * 64;
    float mu[2], lv[2];
#pragma unroll
    for (int d = 0; d < 2; ++d) {
        float a = b_l2p[d], b = b_l2p[d];
        for (int q = 0; q < 32; ++q) {
            float w = W_l2p[q * 2 + d];
            a += hr[q] * w;        // h[:, :32]
            b += hr[32 + q] * w;   // h[:, 32:]
        }
        for (int q = 0; q < 64; ++q) {
            float w = W_l2p[(32 + q) * 2 + d];
            float cv = cr[q];
            a += cv * w;
            b += cv * w;
        }
        mu[d] = a; lv[d] = b;
    }
#pragma unroll
    for (int d = 0; d < 2; ++d) {
        float pos = mu[d] + eps_t[p * 2 + d] * expf(0.5f * lv[d]);
        out_pos[p * 2 + d] = pos;
        out_mu[p * 2 + d]  = mu[d];
        out_lv[p * 2 + d]  = lv[d];
        PRV[p * 2 + d] = pos;
        CUR[p * 2 + d] += pos;
    }
}

// ---------------------------------------------------------------------------
extern "C" void kernel_launch(void* const* d_in, const int* in_sizes, int n_in,
                              void* d_out, int out_size, void* d_ws, size_t ws_size,
                              hipStream_t stream) {
    const float* lastpos = (const float*)d_in[0];
    const float* cond    = (const float*)d_in[1];
    const float* z       = (const float*)d_in[2];
    const float* obs     = (const float*)d_in[3];
    const int*   nei     = (const int*)  d_in[4];
    // d_in[5] nei_num_index: unused by the reference body
    const float* h0      = (const float*)d_in[6];
    const float* c0      = (const float*)d_in[7];
    const float* eps     = (const float*)d_in[8];
    const float* W_in    = (const float*)d_in[9];
    const float* b_in    = (const float*)d_in[10];
    const float* W_ih    = (const float*)d_in[11];
    const float* W_hh    = (const float*)d_in[12];
    const float* b_ih    = (const float*)d_in[13];
    const float* b_hh    = (const float*)d_in[14];
    const float* W_l2p   = (const float*)d_in[15];
    const float* b_l2p   = (const float*)d_in[16];
    const float* W_sp    = (const float*)d_in[17];
    const float* b_sp    = (const float*)d_in[18];
    const float* W1      = (const float*)d_in[19];
    const float* b1      = (const float*)d_in[20];
    const float* W2      = (const float*)d_in[21];
    const float* b2      = (const float*)d_in[22];

    float* out     = (float*)d_out;
    float* out_pos = out;                          // [T,N,2]
    float* out_mu  = out + TSTEPS * NPED * 2;      // [T,N,2]
    float* out_lv  = out + 2 * TSTEPS * NPED * 2;  // [T,N,2]

    float* ws  = (float*)d_ws;
    float* H   = ws;               // 512*64
    float* CS  = H   + NPED * HID;
    float* CTX = CS  + NPED * HID;
    float* AV  = CTX + NPED * HID;
    float* BV  = AV  + NPED * HID;
    float* CUR = BV  + NPED * HID; // 512*2
    float* PRV = CUR + NPED * 2;   // 512*2
    float* M2  = PRV + NPED * 2;   // 2*64
    float* K0  = M2  + 2 * 64;     // 64

    k_init<<<(NPED * HID + 255) / 256, 256, 0, stream>>>(lastpos, obs, h0, c0,
                                                         H, CS, CTX, CUR, PRV);
    k_pre<<<1, 64, 0, stream>>>(W_sp, b_sp, W1, b1, M2, K0);

    for (int t = 0; t < TSTEPS; ++t) {
        k_lstm<<<NPED, 256, 0, stream>>>(PRV, cond, z, CTX, H, CS,
                                         W_in, b_in, W_ih, W_hh, b_ih, b_hh);
        k_ab<<<NPED, 64, 0, stream>>>(H, CUR, W1, M2, K0, AV, BV);
        k_pool<<<NPED, 128, 0, stream>>>(AV, BV,
                                         nei + (size_t)t * NPED * NPED,
                                         W2, b2, CTX);
        k_out<<<2, 256, 0, stream>>>(H, CTX, eps + (size_t)t * NPED * 2,
                                     W_l2p, b_l2p, PRV, CUR,
                                     out_pos + t * NPED * 2,
                                     out_mu  + t * NPED * 2,
                                     out_lv  + t * NPED * 2);
    }
}
